// HardMemory_39204461478031
// MI455X (gfx1250) — matrix operand order repaired and checked by reading, not compiled, not run on hardware
//
#include <hip/hip_runtime.h>

typedef __attribute__((ext_vector_type(16))) _Float16 v16h;
typedef __attribute__((ext_vector_type(8)))  _Float16 v8h;
typedef __attribute__((ext_vector_type(4)))  _Float16 v4h;
typedef __attribute__((ext_vector_type(8)))  float    v8f;
typedef unsigned int u32x4 __attribute__((ext_vector_type(4)));
typedef int          i32x8 __attribute__((ext_vector_type(8)));
typedef int          i32x4 __attribute__((ext_vector_type(4)));

#define C_DIM   128
#define M_ROWS  512
#define HW      4096
#define THRESH  0.8f
#define EPSV    1e-12f
#define TOK_PER_WG 512   // 8 waves * 4 tiles * 16 tokens

// ---------------------------------------------------------------------------
// Kernel 1: L2-normalize the 512x128 memory bank into f16 row-major (d_ws).
// ---------------------------------------------------------------------------
__global__ __launch_bounds__(32)
void hm_norm_mem(const float* __restrict__ mem, _Float16* __restrict__ memh) {
    const int row  = blockIdx.x;
    const int lane = threadIdx.x;
    const float4 v = ((const float4*)(mem + (size_t)row * C_DIM))[lane];
    float s = v.x * v.x + v.y * v.y + v.z * v.z + v.w * v.w;
#pragma unroll
    for (int off = 16; off > 0; off >>= 1) s += __shfl_xor(s, off, 32);
    const float scale = 1.0f / fmaxf(sqrtf(s), EPSV);
    v4h o;
    o[0] = (_Float16)(v.x * scale);
    o[1] = (_Float16)(v.y * scale);
    o[2] = (_Float16)(v.z * scale);
    o[3] = (_Float16)(v.w * scale);
    ((v4h*)(memh + (size_t)row * C_DIM))[lane] = o;
}

// fetch one 16-row x 32-K A tile group (4 K-steps) from the LDS memory bank
__device__ __forceinline__ void load_a(const _Float16* mh, int rowbase, int hi,
                                       v16h a[4]) {
#pragma unroll
    for (int kt = 0; kt < 4; ++kt) {
        const int ab = rowbase + kt * 32 + hi * 8;
        v8h a0 = *(const v8h*)(mh + ab);
        v8h a1 = *(const v8h*)(mh + ab + 16);
        a[kt] = __builtin_shufflevector(a0, a1,
                0,1,2,3,4,5,6,7,8,9,10,11,12,13,14,15);
    }
}

__device__ __forceinline__ void argmax8(const v8f& acc, int base,
                                        float& best, int& bidx) {
#pragma unroll
    for (int r = 0; r < 8; ++r) {
        const float v = acc[r];
        const int idx = base + r;
        if (v > best) { best = v; bidx = idx; }
    }
}

// ---------------------------------------------------------------------------
// Kernel 2: per-token argmax over 512 memory rows via f16 WMMA.
// Memory bank is DMA'd into LDS by the Tensor Data Mover (one issue, wave 0).
// ---------------------------------------------------------------------------
__global__ __launch_bounds__(256)
void hm_main(const float* __restrict__ x,
             const float* __restrict__ mem,
             const _Float16* __restrict__ memh,
             float* __restrict__ out) {
    extern __shared__ char smem[];
    _Float16* mh = (_Float16*)smem;                               // 512*128 f16 = 128 KB
    float* xs = (float*)(smem + M_ROWS * C_DIM * 2)
                + (threadIdx.x >> 5) * (C_DIM * 16);              // 8 KB per wave

    const int tid  = threadIdx.x;
    const int lane = tid & 31;
    const int w    = tid >> 5;
    const int t    = lane & 15;   // token within tile
    const int hi   = lane >> 4;   // half-wave select

    // --- TDM: DMA the whole 512x128 f16 bank from global into LDS ---------
    if (w == 0) {
        const unsigned long long ga = (unsigned long long)memh;   // 57-bit VA
        const unsigned int lds_off = (unsigned int)(unsigned long long)(void*)mh;

        u32x4 g0;
        g0[0] = 1u;                                   // count=1, user descriptor
        g0[1] = lds_off;                              // lds_addr (bytes)
        g0[2] = (unsigned int)(ga & 0xFFFFFFFFull);   // global_addr[31:0]
        g0[3] = (unsigned int)((ga >> 32) & 0x01FFFFFFull)
              | (2u << 30);                           // global_addr[56:32] | type=2

        i32x8 g1;
        g1[0] = 0x10000;              // workgroup_mask=0, data_size=1 (2B)
        g1[1] = (int)(C_DIM << 16);   // tensor_dim0[15:0] at bits 63:48
        g1[2] = (int)(M_ROWS << 16);  // tensor_dim0 hi=0 | tensor_dim1[15:0]
        g1[3] = (int)(C_DIM << 16);   // tensor_dim1 hi=0 | tile_dim0=128
        g1[4] = (int)M_ROWS;          // tile_dim1=512, tile_dim2=0
        g1[5] = (int)C_DIM;           // tensor_dim0_stride[31:0]=128
        g1[6] = 0;                    // stride hi / tensor_dim1_stride lo
        g1[7] = 0;                    // tensor_dim1_stride hi

        i32x4 gz4 = {};               // groups 2/3: zero (2D tensor)
        i32x8 gz8 = {};               // extra group (6-arg toolchain form)
        __builtin_amdgcn_tensor_load_to_lds(g0, g1, gz4, gz4, gz8, 0);
    }

    for (int tile = 0; tile < 4; ++tile) {
        const int tok0 = blockIdx.x * TOK_PER_WG + w * 64 + tile * 16;
        const int b    = tok0 >> 12;          // HW = 4096
        const int n0   = tok0 & (HW - 1);
        const float* xb = x + (size_t)b * C_DIM * HW + n0;

        // stage 16 tokens x 128 channels (channel-major) into LDS, float4 coalesced
#pragma unroll
        for (int j = 0; j < 16; ++j) {
            const int c = j * 8 + (lane >> 2);
            const int f = (lane & 3) * 4;
            float4 v = *(const float4*)(xb + (size_t)c * HW + f);
            *(float4*)(xs + c * 16 + f) = v;
        }
        if (tile == 0 && w == 0) {
            __builtin_amdgcn_s_wait_tensorcnt(0);     // memory bank resident in LDS
        }
        __syncthreads();

        // token L2 norms: half-wave splits channels, combine with shfl_xor 16
        float s = 0.0f;
        const int c0 = hi * 64;
        for (int c = 0; c < 64; ++c) {
            const float v = xs[(c0 + c) * 16 + t];
            s += v * v;
        }
        s += __shfl_xor(s, 16, 32);
        const float scale = 1.0f / fmaxf(sqrtf(s), EPSV);

        // build B operand (K x 16 tokens): lane holds K {8h..8h+7, 16+8h..16+8h+7} of token t
        v16h bR[4];
#pragma unroll
        for (int kt = 0; kt < 4; ++kt) {
            const int kb = kt * 32 + hi * 8;
#pragma unroll
            for (int j = 0; j < 16; ++j)
                bR[kt][j] = (_Float16)(xs[(kb + (j & 7) + ((j >> 3) << 4)) * 16 + t] * scale);
        }

        // sweep all 512 memory rows, 16 per WMMA tile, double-buffered A fetch
        float best = -1e30f;
        int   bidx = 0;
        v16h aA[4], aB[4];
        load_a(mh, (0 * 16 + t) * C_DIM, hi, aA);
#pragma unroll 1
        for (int m = 0; m < 32; m += 2) {
            // prefetch tile m+1 while computing tile m
            load_a(mh, ((m + 1) * 16 + t) * C_DIM, hi, aB);
            v8f acc = {};
#pragma unroll
            for (int kt = 0; kt < 4; ++kt)
                acc = __builtin_amdgcn_wmma_f32_16x16x32_f16(
                        false, aA[kt], false, bR[kt], (short)0, acc, false, false);
            argmax8(acc, m * 16 + hi * 8, best, bidx);

            // prefetch tile m+2 (wraps to 0 on last iter: harmless) while computing m+1
            load_a(mh, (((m + 2) & 31) * 16 + t) * C_DIM, hi, aA);
            v8f acc2 = {};
#pragma unroll
            for (int kt = 0; kt < 4; ++kt)
                acc2 = __builtin_amdgcn_wmma_f32_16x16x32_f16(
                        false, aB[kt], false, bR[kt], (short)0, acc2, false, false);
            argmax8(acc2, (m + 1) * 16 + hi * 8, best, bidx);
        }
        // merge the two half-wave row ranges (first-max tie-break on lower idx)
        {
            const float ov = __shfl_xor(best, 16, 32);
            const int   oi = __shfl_xor(bidx, 16, 32);
            if (ov > best || (ov == best && oi < bidx)) { best = ov; bidx = oi; }
        }

        // write output: selected un-normalized memory row (L2-hot gather) or zeros
        const bool on = best > THRESH;
        const float* mrow = mem + (size_t)bidx * C_DIM;
        float* ob = out + (size_t)b * C_DIM * HW + n0;
#pragma unroll 8
        for (int j = 0; j < 64; ++j) {
            const int c = j * 2 + hi;
            const float v = on ? mrow[c] : 0.0f;
            ob[(size_t)c * HW + t] = v;
        }
        __syncthreads();
    }
}

// ---------------------------------------------------------------------------
extern "C" void kernel_launch(void* const* d_in, const int* in_sizes, int n_in,
                              void* d_out, int out_size, void* d_ws, size_t ws_size,
                              hipStream_t stream) {
    (void)in_sizes; (void)n_in; (void)out_size; (void)ws_size;
    const float* x    = (const float*)d_in[0];   // [32,128,64,64] f32
    const float* memf = (const float*)d_in[1];   // [512,128] f32
    float*       outp = (float*)d_out;           // [32,128,64,64] f32
    _Float16*    memh = (_Float16*)d_ws;         // 512*128 f16 = 128 KB scratch

    hm_norm_mem<<<M_ROWS, 32, 0, stream>>>(memf, memh);

    const int n_tokens = 32 * HW;                       // 131072
    const int grid = n_tokens / TOK_PER_WG;             // 256 workgroups
    const size_t shmem = (size_t)M_ROWS * C_DIM * 2     // 128 KB memory bank
                       + (size_t)8 * C_DIM * 16 * 4;    // 64 KB x staging (8 waves)
    hm_main<<<grid, 256, shmem, stream>>>(x, memf, memh, outp);
}
